// Attention_55499567399068
// MI455X (gfx1250) — matrix operand order repaired and checked by reading, not compiled, not run on hardware
//
#include <hip/hip_runtime.h>

typedef _Float16 h16 __attribute__((ext_vector_type(16)));
typedef _Float16 h8  __attribute__((ext_vector_type(8)));
typedef _Float16 h4  __attribute__((ext_vector_type(4)));
typedef float    f8  __attribute__((ext_vector_type(8)));

#define DD   512   // feature dim
#define NS   64    // nshot (rows per (b,n) tile)
#define LDSB (4*65536 + 16384 + 256)

// ---------------- WMMA helpers (CDNA5 wave32, 16x16x32 f16 -> f32) -------------

__device__ __forceinline__ f8 wmma16(h16 a, h16 b, f8 c) {
    // D = A x B + C ; emits v_wmma_f32_16x16x32_f16
    return __builtin_amdgcn_wmma_f32_16x16x32_f16(false, a, false, b, (short)0, c,
                                                  false, false);
}

// A-fragment: 16x32 f16, row-major source with leading dim `ld`.
// ISA layout: lanes 0-15 hold K=0..7 & 16..23 ; lanes 16-31 hold K=8..15 & 24..31.
__device__ __forceinline__ h16 load_a(const _Float16* base, int ld, int m0, int k0,
                                      int lane) {
    const int m    = m0 + (lane & 15);
    const int ksel = (lane >> 4) * 8;
    const _Float16* p = base + m * ld + k0;
    h8 lo = *(const h8*)(p + ksel);
    h8 hi = *(const h8*)(p + 16 + ksel);
    h16 a;
#pragma unroll
    for (int i = 0; i < 8; ++i) { a[i] = lo[i]; a[i + 8] = hi[i]; }
    return a;
}

// B-fragment: 32x16 f16 (KxN), row-major source (row = K) with leading dim `ld`.
// ISA layout: lane L holds column N = n0+L%16; lanes 0-15 hold K=0..7 & 16..23 ; lanes 16-31 hold K=8..15 & 24..31.
__device__ __forceinline__ h16 load_b(const _Float16* base, int ld, int k0, int n0,
                                      int lane) {
    const _Float16* p = base + (k0 + (lane >> 4) * 8) * ld + n0 + (lane & 15);
    h16 b;
#pragma unroll
    for (int i = 0; i < 8; ++i) { b[i] = p[i * ld]; b[i + 8] = p[(16 + i) * ld]; }
    return b;
}

// 4x4-blocked projection GEMM: dst = src(64x512) @ WT(512x512) + bias.
// Wave `wave` exclusively owns n-tiles [wave*4, wave*4+4) across all 4 m-slabs,
// so every weight fragment is fetched exactly once per workgroup.
__device__ __forceinline__ void proj_gemm(const _Float16* __restrict__ src,
                                          const _Float16* __restrict__ WT,
                                          const float* __restrict__ bias,
                                          _Float16* __restrict__ dst,
                                          int wave, int lane, bool transposed) {
    const int g = wave;                       // n-tile group
    f8 c[4][4] = {};
    for (int kk = 0; kk < 16; ++kk) {
        h16 a[4], b[4];
#pragma unroll
        for (int m = 0; m < 4; ++m) a[m] = load_a(src, DD, m * 16, kk * 32, lane);
#pragma unroll
        for (int j = 0; j < 4; ++j)
            b[j] = load_b(WT, DD, kk * 32, (g * 4 + j) * 16, lane);
#pragma unroll
        for (int m = 0; m < 4; ++m)
#pragma unroll
            for (int j = 0; j < 4; ++j) c[m][j] = wmma16(a[m], b[j], c[m][j]);
    }
#pragma unroll
    for (int m = 0; m < 4; ++m)
#pragma unroll
        for (int j = 0; j < 4; ++j) {
            const int n = (g * 4 + j) * 16 + (lane & 15);
            const float bb = bias[n];
            const int mr = m * 16 + 8 * (lane >> 4);
            if (!transposed) {
#pragma unroll
                for (int r = 0; r < 8; ++r)
                    dst[(mr + r) * DD + n] = (_Float16)(c[m][j][r] + bb);
            } else {                           // dst is h-major 512x64 (kT)
                h8 pk;
#pragma unroll
                for (int r = 0; r < 8; ++r) pk[r] = (_Float16)(c[m][j][r] + bb);
                *(h8*)(dst + n * NS + mr) = pk;
            }
        }
}

// ------------------------------ prep: weights -> f16, transposed ---------------
// ws layout (f16 elems): WqT[0], WkT[262144], WvT[524288], Wr1T[786432]
__global__ __launch_bounds__(256) void prep_weights(const float* __restrict__ Wq,
                                                    const float* __restrict__ Wk,
                                                    const float* __restrict__ Wv,
                                                    const float* __restrict__ Wr1,
                                                    _Float16* __restrict__ ws) {
    const int tid = blockIdx.x * 256 + threadIdx.x;   // 0 .. 262143
    const int d = tid >> 9, h = tid & 511;
    ws[tid]          = (_Float16)Wq[h * 512 + d];     // WqT[d][h]
    ws[262144 + tid] = (_Float16)Wk[h * 512 + d];
    ws[524288 + tid] = (_Float16)Wv[h * 512 + d];
    if (tid < 32768) {                                 // Wr1T[h][m], 512x64
        const int hh = tid >> 6, m = tid & 63;
        ws[786432 + tid] = (_Float16)Wr1[m * 512 + hh];
    }
}

// ------------------------------ fused attention kernel -------------------------
__global__ __launch_bounds__(256) void attn_fused(
    const float* __restrict__ Q, const float* __restrict__ K,
    const float* __restrict__ V, const _Float16* __restrict__ ws,
    const float* __restrict__ bq, const float* __restrict__ bk,
    const float* __restrict__ bv, const float* __restrict__ br1,
    const float* __restrict__ Wr2, const float* __restrict__ br2,
    float* __restrict__ out) {
    const int bn   = blockIdx.x;      // 0..2047 = b*8+n
    const int tid  = threadIdx.x;
    const int wave = tid >> 5, lane = tid & 31;

    extern __shared__ char smem[];
    _Float16* sA = (_Float16*)smem;       // rawQ  -> kT (512x64, ld 64)
    _Float16* sB = sA + NS * DD;          // rawK  -> v   (64x512)
    _Float16* sC = sB + NS * DD;          // rawV  -> context (64x512)
    _Float16* sD = sC + NS * DD;          // q (64x512) -> att (64x64, ld 64)
    float*    sE = (float*)(sD + NS * DD);// scores -> hid (64x64 f32)
    float*    sW = sE + NS * NS;          // w[64]

    const _Float16* WqT  = ws;
    const _Float16* WkT  = ws + 262144;
    const _Float16* WvT  = ws + 524288;
    const _Float16* Wr1T = ws + 786432;

    // ---- S1: stage raw Q/K/V tiles into LDS as f16 (coalesced 16B loads) ----
    const size_t base = (size_t)bn * NS * DD;
    {
        const float4* q4 = (const float4*)(Q + base);
        const float4* k4 = (const float4*)(K + base);
        const float4* v4 = (const float4*)(V + base);
#pragma unroll 4
        for (int i = 0; i < 32; ++i) {
            const int idx = i * 256 + tid;           // 8192 float4 per tensor
            float4 f = q4[idx];
            *(h4*)(sA + idx * 4) = h4{(_Float16)f.x, (_Float16)f.y,
                                      (_Float16)f.z, (_Float16)f.w};
            f = k4[idx];
            *(h4*)(sB + idx * 4) = h4{(_Float16)f.x, (_Float16)f.y,
                                      (_Float16)f.z, (_Float16)f.w};
            f = v4[idx];
            *(h4*)(sC + idx * 4) = h4{(_Float16)f.x, (_Float16)f.y,
                                      (_Float16)f.z, (_Float16)f.w};
        }
    }
    __syncthreads();

    const int nlane = lane & 15;
    const int mlane = 8 * (lane >> 4);

    // ---- S2: q = rawQ @ WqT + bq  -> sD (row-major, ld 512) ----
    proj_gemm(sA, WqT, bq, sD, wave, lane, false);
    __syncthreads();

    // ---- S3: kT = (rawK @ WkT + bk)^T -> sA (h-major 512x64, ld 64) ----
    proj_gemm(sB, WkT, bk, sA, wave, lane, true);
    __syncthreads();

    // ---- S4: v = rawV @ WvT + bv -> sB (row-major, ld 512) ----
    proj_gemm(sC, WvT, bv, sB, wave, lane, false);
    __syncthreads();

    // ---- S5: scores = q @ kT / sqrt(512) -> sE (64x64 f32) ----
    for (int i = 0; i < 2; ++i) {
        const int t = wave * 2 + i, smt = t >> 2, snt = t & 3;
        f8 c = {};
        for (int kk = 0; kk < 16; ++kk)
            c = wmma16(load_a(sD, DD, smt * 16, kk * 32, lane),
                       load_b(sA, NS, kk * 32, snt * 16, lane), c);
        const int n = snt * 16 + nlane, mr = smt * 16 + mlane;
#pragma unroll
        for (int r = 0; r < 8; ++r)
            sE[(mr + r) * NS + n] = c[r] * 0.04419417382415922f; // 1/sqrt(512)
    }
    __syncthreads();

    // ---- S6: softmax rows -> att f16 in sD (ld 64) ----
    if (tid < NS) {
        float mx = -3.4e38f;
        for (int j = 0; j < NS; ++j) mx = fmaxf(mx, sE[tid * NS + j]);
        float s = 0.f;
        for (int j = 0; j < NS; ++j) {
            const float e = __expf(sE[tid * NS + j] - mx);
            sE[tid * NS + j] = e;
            s += e;
        }
        const float inv = 1.f / s;
        for (int j = 0; j < NS; ++j)
            sD[tid * NS + j] = (_Float16)(sE[tid * NS + j] * inv);
    }
    __syncthreads();

    // ---- S7: context = att @ v -> sC (64x512 f16), 4x4 blocked ----
    {
        f8 c[4][4] = {};
        for (int kk = 0; kk < 2; ++kk) {
            h16 a[4], b[4];
#pragma unroll
            for (int m = 0; m < 4; ++m)
                a[m] = load_a(sD, NS, m * 16, kk * 32, lane);
#pragma unroll
            for (int j = 0; j < 4; ++j)
                b[j] = load_b(sB, DD, kk * 32, (wave * 4 + j) * 16, lane);
#pragma unroll
            for (int m = 0; m < 4; ++m)
#pragma unroll
                for (int j = 0; j < 4; ++j) c[m][j] = wmma16(a[m], b[j], c[m][j]);
        }
#pragma unroll
        for (int m = 0; m < 4; ++m)
#pragma unroll
            for (int j = 0; j < 4; ++j) {
                const int n = (wave * 4 + j) * 16 + nlane;
                const int mr = m * 16 + mlane;
#pragma unroll
                for (int r = 0; r < 8; ++r)
                    sC[(mr + r) * DD + n] = (_Float16)c[m][j][r];
            }
    }
    __syncthreads();

    // ---- S8: hid = leaky(context @ Wr1T + br1) -> sE (64x64 f32) ----
    for (int i = 0; i < 2; ++i) {
        const int t = wave * 2 + i, smt = t >> 2, snt = t & 3;
        f8 c = {};
        for (int kk = 0; kk < 16; ++kk)
            c = wmma16(load_a(sC, DD, smt * 16, kk * 32, lane),
                       load_b(Wr1T, NS, kk * 32, snt * 16, lane), c);
        const int n = snt * 16 + nlane, mr = smt * 16 + mlane;
        const float bb = br1[n];
#pragma unroll
        for (int r = 0; r < 8; ++r) {
            const float x = c[r] + bb;
            sE[(mr + r) * NS + n] = (x >= 0.f) ? x : 0.01f * x;
        }
    }
    __syncthreads();

    // ---- S9: w[q] = hid[q,:] . Wr2 + br2 ----
    if (tid < NS) {
        float acc = br2[0];
        for (int m = 0; m < NS; ++m) acc += sE[tid * NS + m] * Wr2[m];
        sW[tid] = acc;
    }
    __syncthreads();

    // ---- S10: out[h] = sum_q context[q,h] * w[q] ----
    for (int h = tid; h < DD; h += 256) {
        float acc = 0.f;
        for (int q = 0; q < NS; ++q) acc += (float)sC[q * DD + h] * sW[q];
        out[(size_t)bn * DD + h] = acc;
    }
}

// ------------------------------------ host -------------------------------------
extern "C" void kernel_launch(void* const* d_in, const int* in_sizes, int n_in,
                              void* d_out, int out_size, void* d_ws, size_t ws_size,
                              hipStream_t stream) {
    const float* Q   = (const float*)d_in[0];
    const float* K   = (const float*)d_in[1];
    const float* V   = (const float*)d_in[2];
    const float* Wq  = (const float*)d_in[3];
    const float* bq  = (const float*)d_in[4];
    const float* Wk  = (const float*)d_in[5];
    const float* bk  = (const float*)d_in[6];
    const float* Wv  = (const float*)d_in[7];
    const float* bv  = (const float*)d_in[8];
    const float* Wr1 = (const float*)d_in[9];
    const float* br1 = (const float*)d_in[10];
    const float* Wr2 = (const float*)d_in[11];
    const float* br2 = (const float*)d_in[12];
    float* out = (float*)d_out;
    _Float16* ws = (_Float16*)d_ws;   // 1.57 MB used: WqT/WkT/WvT/Wr1T in f16

    (void)hipFuncSetAttribute((const void*)attn_fused,
                              hipFuncAttributeMaxDynamicSharedMemorySize, LDSB);

    prep_weights<<<1024, 256, 0, stream>>>(Wq, Wk, Wv, Wr1, ws);
    attn_fused<<<2048, 256, LDSB, stream>>>(Q, K, V, ws, bq, bk, bv, br1, Wr2, br2,
                                            out);
}
